// UniAnchorGNN_25838523253004
// MI455X (gfx1250) — compile-verified, matrix operand order checked by reading, not run on hardware
//
#include <hip/hip_runtime.h>

#define M_ 4
#define B_ 256
#define NPG_ 64
#define N_ (B_*NPG_)        // 16384
#define DEG_ 8
#define E_ (N_*DEG_)        // 131072
#define DF_ 16
#define D_ 256
#define L_ 5
#define NA_ 2
#define TASKS_ 10
#define EPG_ (NPG_*DEG_)    // 512 edges per graph
#define AS_STRIDE 264       // 256 + 8 bf16 pad (16B) -> conflict-free ds_load_b128

typedef __bf16 bf16;
typedef __attribute__((ext_vector_type(16))) __bf16 v16bf;
typedef __attribute__((ext_vector_type(8)))  __bf16 v8bf;
typedef __attribute__((ext_vector_type(8)))  float  v8f;
typedef __attribute__((ext_vector_type(4)))  float  f32x4;

// ---------------------------------------------------------------------------
// WMMA fragment load: 16-bit A/B fragment for 16x16x32.
// lane l (l<16): row = l, K chunks [kb + 0..7] and [kb+16..23]
// lane l (l>=16): row = l-16, K chunks [kb + 8..15] and [kb+24..31]
// -> two contiguous 16B loads per lane at kb + sel*8 and kb + 16 + sel*8.
// ---------------------------------------------------------------------------
__device__ __forceinline__ v16bf frag_ld(const bf16* rowptr, int kb, int sel) {
  union { v16bf v; f32x4 q[2]; } u;
  u.q[0] = *(const f32x4*)(rowptr + kb + sel * 8);
  u.q[1] = *(const f32x4*)(rowptr + kb + 16 + sel * 8);
  return u.v;
}

// ---------------------------------------------------------------------------
// Fused (optional edge-aggregation) + GEMM + bias + ReLU, bf16 in/out.
// One workgroup = 64 node rows (one (m, graph) tile) x full 256 cols, K=256.
// Wave tiling: 8 waves, wave w covers all 64 rows x 32-col window (w*32).
// -> weight matrix is read from L2 exactly once per workgroup.
// Aggregation: deterministic counting-sort of edges by dst, register gather.
// ---------------------------------------------------------------------------
__global__ void __launch_bounds__(256)
gemm_kernel(const bf16* __restrict__ Ain, bf16* __restrict__ Out,
            const bf16* __restrict__ WT, const float* __restrict__ bias,
            const float* __restrict__ eps_arr, int layer,
            const int* __restrict__ esrc, const int* __restrict__ edst,
            int do_agg)
{
  extern __shared__ char smem_raw[];
  bf16* As   = (bf16*)smem_raw;                 // 64 x AS_STRIDE bf16 (33792 B)
  int*  Ess  = (int*)(smem_raw + 64 * AS_STRIDE * 2);  // 512
  int*  Esd  = Ess + EPG_;                      // 512
  int*  order= Esd + EPG_;                      // 512 (src ids sorted by dst)
  int*  cnt  = order + EPG_;                    // 64
  int*  offs = cnt + 64;                        // 65

  const int tid = threadIdx.x;
  const int bid = blockIdx.x;                 // 0..1023 : m = bid>>8, b = bid&255
  const size_t rowbase = (size_t)bid * 64;    // == m*N_ + b*64

  // ---- load 64x256 A tile into LDS ----
  {
    const int r = tid >> 2, q = tid & 3;
    const bf16* src = Ain + (rowbase + r) * D_ + q * 64;
    bf16* dst = As + r * AS_STRIDE + q * 64;
    #pragma unroll
    for (int c = 0; c < 64; c += 8)
      *(f32x4*)(dst + c) = *(const f32x4*)(src + c);
  }

  if (do_agg) {
    const int b  = bid & (B_ - 1);
    const int eb = b * EPG_;
    if (tid < 64) cnt[tid] = 0;
    for (int e = tid; e < EPG_; e += 256) {
      Ess[e] = esrc[eb + e] - b * NPG_;
      Esd[e] = edst[eb + e] - b * NPG_;
    }
    __syncthreads();
    // histogram of destinations (integer atomics: deterministic)
    for (int e = tid; e < EPG_; e += 256) atomicAdd(&cnt[Esd[e]], 1);
    __syncthreads();
    if (tid == 0) {
      int s = 0;
      for (int i = 0; i < 64; i++) { offs[i] = s; s += cnt[i]; }
      offs[64] = s;
    }
    __syncthreads();
    // stable scatter: thread i (<64) collects edges with dst==i in index order
    if (tid < 64) {
      int w = offs[tid];
      for (int e = 0; e < EPG_; e++)
        if (Esd[e] == tid) order[w++] = Ess[e];
    }
    __syncthreads();
    // gather: thread handles (dst = tid>>2, 64-dim quarter q), edge-index order
    {
      const float sc = 1.0f + eps_arr[layer];
      const int dr = tid >> 2, q = tid & 3;
      float reg[64];
      const bf16* hrow = As + dr * AS_STRIDE + q * 64;
      #pragma unroll
      for (int c = 0; c < 64; c += 8) {
        v8bf hb = __builtin_bit_cast(v8bf, *(const f32x4*)(hrow + c));
        #pragma unroll
        for (int j = 0; j < 8; j++) reg[c + j] = sc * (float)hb[j];
      }
      const int k0 = offs[dr], k1 = offs[dr + 1];
      for (int k = k0; k < k1; k++) {
        const bf16* sp = As + order[k] * AS_STRIDE + q * 64;
        #pragma unroll
        for (int c = 0; c < 64; c += 8) {
          v8bf sb = __builtin_bit_cast(v8bf, *(const f32x4*)(sp + c));
          #pragma unroll
          for (int j = 0; j < 8; j++) reg[c + j] += (float)sb[j];
        }
      }
      __syncthreads();   // all reads of As done
      bf16* wrow = As + dr * AS_STRIDE + q * 64;
      #pragma unroll
      for (int c = 0; c < 64; c += 8) {
        v8bf ob;
        #pragma unroll
        for (int j = 0; j < 8; j++) ob[j] = (bf16)reg[c + j];
        *(f32x4*)(wrow + c) = __builtin_bit_cast(f32x4, ob);
      }
    }
  }
  __syncthreads();

  // ---- WMMA GEMM: wave w -> cols [w*32, w*32+32), all 64 rows ----
  const int wid = tid >> 5, lane = tid & 31;
  const int ncol = wid * 32;
  const int nl = lane & 15, sel = lane >> 4;

  v8f c[4][2] = {};

  for (int kb = 0; kb < 256; kb += 32) {
    v16bf a[4], bfr[2];
    #pragma unroll
    for (int mi = 0; mi < 4; mi++)
      a[mi] = frag_ld(As + (mi * 16 + nl) * AS_STRIDE, kb, sel);
    #pragma unroll
    for (int ni = 0; ni < 2; ni++)
      bfr[ni] = frag_ld(WT + (size_t)(ncol + ni * 16 + nl) * D_, kb, sel);
    #pragma unroll
    for (int mi = 0; mi < 4; mi++)
      #pragma unroll
      for (int ni = 0; ni < 2; ni++)
        c[mi][ni] = __builtin_amdgcn_wmma_f32_16x16x32_bf16(
            false, a[mi], false, bfr[ni], (short)0, c[mi][ni], false, false);
  }

  __syncthreads();   // done reading As; reuse it to stage C for coalesced store

  // C layout: lane<16 -> col = nl, rows r (VGPR r); lane>=16 -> rows r+8.
  #pragma unroll
  for (int ni = 0; ni < 2; ni++) {
    const int n = ncol + ni * 16 + nl;
    const float bv = bias[n];
    #pragma unroll
    for (int mi = 0; mi < 4; mi++) {
      #pragma unroll
      for (int r = 0; r < 8; r++) {
        const int row = mi * 16 + sel * 8 + r;
        As[row * AS_STRIDE + n] = (bf16)fmaxf(c[mi][ni][r] + bv, 0.0f);
      }
    }
  }
  __syncthreads();

  // coalesced b128 stores
  {
    const int r = tid >> 2, q = tid & 3;
    const bf16* srow = As + r * AS_STRIDE + q * 64;
    bf16* drow = Out + (rowbase + r) * D_ + q * 64;
    #pragma unroll
    for (int c2 = 0; c2 < 64; c2 += 8)
      *(f32x4*)(drow + c2) = *(const f32x4*)(srow + c2);
  }
}

// ---------------------------------------------------------------------------
// Per-graph pairwise min-distance (64 nodes x 256 dims). Deterministic.
// ---------------------------------------------------------------------------
__global__ void __launch_bounds__(256)
mindist_kernel(const bf16* __restrict__ H, float* __restrict__ mind)
{
  extern __shared__ char smem_raw[];
  bf16* Hs   = (bf16*)smem_raw;                          // 64 x AS_STRIDE
  float* sq  = (float*)(smem_raw + 64 * AS_STRIDE * 2);  // 64
  float* sqp = sq + 64;                                  // 256 partials
  float* pm  = sqp + 256;                                // 256 partial mins

  const int tid = threadIdx.x, bid = blockIdx.x;
  const size_t rowbase = (size_t)bid * 64;
  const int r = tid >> 2, q = tid & 3;

  {
    const bf16* src = H + (rowbase + r) * D_ + q * 64;
    bf16* dst = Hs + r * AS_STRIDE + q * 64;
    float s = 0.0f;
    #pragma unroll
    for (int c = 0; c < 64; c += 8) {
      f32x4 v = *(const f32x4*)(src + c);
      *(f32x4*)(dst + c) = v;
      v8bf e = __builtin_bit_cast(v8bf, v);
      #pragma unroll
      for (int j = 0; j < 8; j++) { float f = (float)e[j]; s += f * f; }
    }
    sqp[tid] = s;
  }
  __syncthreads();
  if (tid < 64)
    sq[tid] = (sqp[tid * 4] + sqp[tid * 4 + 1]) + (sqp[tid * 4 + 2] + sqp[tid * 4 + 3]);
  __syncthreads();
  {
    float mn = 1e30f;
    const bf16* hi = Hs + r * AS_STRIDE;
    for (int j = q * 16; j < q * 16 + 16; j++) {
      const bf16* hj = Hs + j * AS_STRIDE;
      float dot = 0.0f;
      for (int k = 0; k < 256; k += 8) {
        v8bf ea = __builtin_bit_cast(v8bf, *(const f32x4*)(hi + k));
        v8bf eb = __builtin_bit_cast(v8bf, *(const f32x4*)(hj + k));
        #pragma unroll
        for (int u = 0; u < 8; u++) dot += (float)ea[u] * (float)eb[u];
      }
      float d2 = fmaxf(sq[r] + sq[j] - 2.0f * dot, 0.0f);
      if (j == r) d2 += 1e9f;
      mn = fminf(mn, d2);
    }
    pm[r * 4 + q] = mn;
  }
  __syncthreads();
  if (tid < 64) {
    float mn = fminf(fminf(pm[tid * 4], pm[tid * 4 + 1]),
                     fminf(pm[tid * 4 + 2], pm[tid * 4 + 3]));
    mind[rowbase + tid] = mn;
  }
}

// ---------------------------------------------------------------------------
// threefry-2x32-20 (matches JAX threefry_2x32).
// ---------------------------------------------------------------------------
__device__ __forceinline__ void threefry2x32(unsigned k0, unsigned k1,
                                             unsigned c0, unsigned c1,
                                             unsigned* o0, unsigned* o1)
{
  const unsigned ks2 = 0x1BD11BDAu ^ k0 ^ k1;
  unsigned x0 = c0 + k0, x1 = c1 + k1;
  const int R0[4] = {13, 15, 26, 6}, R1[4] = {17, 29, 16, 24};
#define RR_(x, n) (((x) << (n)) | ((x) >> (32 - (n))))
  #pragma unroll
  for (int i = 0; i < 4; i++) { x0 += x1; x1 = RR_(x1, R0[i]); x1 ^= x0; }
  x0 += k1;  x1 += ks2 + 1u;
  #pragma unroll
  for (int i = 0; i < 4; i++) { x0 += x1; x1 = RR_(x1, R1[i]); x1 ^= x0; }
  x0 += ks2; x1 += k0 + 2u;
  #pragma unroll
  for (int i = 0; i < 4; i++) { x0 += x1; x1 = RR_(x1, R0[i]); x1 ^= x0; }
  x0 += k0;  x1 += k1 + 3u;
  #pragma unroll
  for (int i = 0; i < 4; i++) { x0 += x1; x1 = RR_(x1, R1[i]); x1 ^= x0; }
  x0 += k1;  x1 += ks2 + 4u;
  #pragma unroll
  for (int i = 0; i < 4; i++) { x0 += x1; x1 = RR_(x1, R0[i]); x1 ^= x0; }
  x0 += ks2; x1 += k0 + 5u;
#undef RR_
  *o0 = x0; *o1 = x1;
}

// Gumbel-argmax anchor selection per (m, graph). One block of 64 threads.
__global__ void __launch_bounds__(64)
anchor_kernel(const float* __restrict__ mind, int* __restrict__ anchor, int iter)
{
  __shared__ float zs[64];
  const int p = threadIdx.x;
  const int bid = blockIdx.x;               // m*B_ + b
  const int m = bid >> 8, b = bid & 255;

  unsigned k0, k1;
  threefry2x32(0u, 42u, 0u, (unsigned)iter, &k0, &k1);   // fold_in(key(42), iter)

  const unsigned idx  = (unsigned)(bid * 64 + p);        // flat (m,b,p)
  const unsigned half = (M_ * B_ * NPG_) / 2;
  unsigned o0, o1, bits;
  if (idx < half) { threefry2x32(k0, k1, idx, idx + half, &o0, &o1); bits = o0; }
  else            { threefry2x32(k0, k1, idx - half, idx, &o0, &o1); bits = o1; }

  unsigned fb = (bits >> 9) | 0x3F800000u;
  float f = __builtin_bit_cast(float, fb);
  float u = fmaxf((f - 1.0f) * (1.0f - 1e-9f) + 1e-9f, 1e-9f);

  const int node = b * 64 + p;
  const int anc = anchor[m * N_ + node];
  const float pred = -mind[m * N_ + node] - 10.0f * ((anc > 0) ? 1.0f : 0.0f);
  zs[p] = pred - logf(-logf(u));
  __syncthreads();
  if (p == 0) {
    int best = 0; float bz = zs[0];
    for (int j = 1; j < 64; j++) if (zs[j] > bz) { bz = zs[j]; best = j; }
    anchor[m * N_ + b * 64 + best] = iter;
  }
}

// xcur[m,n,:] = h0[n,:] * (1 + anchor_emb[anchor[m,n],:])
__global__ void __launch_bounds__(256)
xcur_kernel(const bf16* __restrict__ h0, const int* __restrict__ anchor,
            const float* __restrict__ aemb, bf16* __restrict__ Out)
{
  const int bid = blockIdx.x;        // m*N_ + n  (0..65535)
  const int d = threadIdx.x;
  const int n = bid & (N_ - 1);
  const int a = anchor[bid];
  const float t = (float)h0[(size_t)n * D_ + d];
  Out[(size_t)bid * D_ + d] = (bf16)(t * (1.0f + aemb[a * D_ + d]));
}

// h0 = relu(x @ W_enc + b_enc), bf16 out
__global__ void __launch_bounds__(256)
encode_kernel(const float* __restrict__ x, const float* __restrict__ Wenc,
              const float* __restrict__ benc, bf16* __restrict__ h0)
{
  __shared__ float xs[DF_];
  const int n = blockIdx.x, d = threadIdx.x;
  if (d < DF_) xs[d] = x[n * DF_ + d];
  __syncthreads();
  float s = benc[d];
  #pragma unroll
  for (int k = 0; k < DF_; k++) s += xs[k] * Wenc[k * D_ + d];
  h0[(size_t)n * D_ + d] = (bf16)fmaxf(s, 0.0f);
}

// broadcast h0 to all M replicas (initial xcur = tx)
__global__ void __launch_bounds__(256)
init_kernel(const bf16* __restrict__ h0, bf16* __restrict__ Out)
{
  const int n = blockIdx.x, d = threadIdx.x;
  const bf16 v = h0[(size_t)n * D_ + d];
  #pragma unroll
  for (int m = 0; m < M_; m++) Out[((size_t)m * N_ + n) * D_ + d] = v;
}

__global__ void zero_i32(int* __restrict__ p, int n)
{
  const int i = blockIdx.x * 256 + threadIdx.x;
  if (i < n) p[i] = 0;
}

// WT[n,k] = bf16(W[k,n])
__global__ void __launch_bounds__(256)
wt_kernel(const float* __restrict__ W, bf16* __restrict__ WT)
{
  const int n = blockIdx.x, k = threadIdx.x;
  WT[(size_t)n * D_ + k] = (bf16)W[(size_t)k * D_ + n];
}

// hg[m,b,d] = mean_p hn[m, b*64+p, d]
__global__ void __launch_bounds__(256)
pool_kernel(const bf16* __restrict__ Hn, float* __restrict__ hg)
{
  const int bid = blockIdx.x;   // m*B_ + b
  const int d = threadIdx.x;
  const bf16* base = Hn + (size_t)bid * 64 * D_ + d;
  float s = 0.0f;
  #pragma unroll
  for (int p = 0; p < 64; p++) s += (float)base[(size_t)p * D_];
  hg[(size_t)bid * D_ + d] = s * (1.0f / 64.0f);
}

// out[b,t] = mean_m (hg[m,b,:] @ W_pred[:,t]) + b_pred[t]
__global__ void __launch_bounds__(64)
head_kernel(const float* __restrict__ hg, const float* __restrict__ Wp,
            const float* __restrict__ bp, float* __restrict__ out)
{
  const int b = blockIdx.x, t = threadIdx.x;
  if (t >= TASKS_) return;
  float s = 0.0f;
  for (int m = 0; m < M_; m++) {
    const float* h = hg + (size_t)(m * B_ + b) * D_;
    float a = 0.0f;
    for (int d = 0; d < D_; d++) a += h[d] * Wp[d * TASKS_ + t];
    s += a;
  }
  out[b * TASKS_ + t] = s * (1.0f / M_) + bp[t];
}

// ---------------------------------------------------------------------------
extern "C" void kernel_launch(void* const* d_in, const int* in_sizes, int n_in,
                              void* d_out, int out_size, void* d_ws, size_t ws_size,
                              hipStream_t stream)
{
  const float* x     = (const float*)d_in[0];
  const float* W_enc = (const float*)d_in[1];
  const float* b_enc = (const float*)d_in[2];
  const float* gW1   = (const float*)d_in[3];
  const float* gb1   = (const float*)d_in[4];
  const float* gW2   = (const float*)d_in[5];
  const float* gb2   = (const float*)d_in[6];
  const float* eps   = (const float*)d_in[7];
  const float* aemb  = (const float*)d_in[8];
  const float* Wn    = (const float*)d_in[9];
  const float* bn    = (const float*)d_in[10];
  const float* Wp    = (const float*)d_in[11];
  const float* bp    = (const float*)d_in[12];
  const int*   esrc  = (const int*)d_in[13];
  const int*   edst  = (const int*)d_in[14];
  float* out = (float*)d_out;

  char* ws = (char*)d_ws;
  size_t off = 0;
  auto alloc = [&](size_t bytes) {
    void* p = ws + off;
    off += (bytes + 255) & ~(size_t)255;
    return p;
  };
  bf16*  h0     = (bf16*)alloc((size_t)N_ * D_ * 2);
  bf16*  bufA   = (bf16*)alloc((size_t)M_ * N_ * D_ * 2);
  bf16*  bufB   = (bf16*)alloc((size_t)M_ * N_ * D_ * 2);
  bf16*  W1T    = (bf16*)alloc((size_t)L_ * D_ * D_ * 2);
  bf16*  W2T    = (bf16*)alloc((size_t)L_ * D_ * D_ * 2);
  bf16*  WnT    = (bf16*)alloc((size_t)D_ * D_ * 2);
  float* mind   = (float*)alloc((size_t)M_ * N_ * 4);
  int*   anchor = (int*)alloc((size_t)M_ * N_ * 4);
  float* hg     = (float*)alloc((size_t)M_ * B_ * D_ * 4);
  (void)ws_size; (void)in_sizes; (void)n_in; (void)out_size;

  // weight prep: bf16 transposed copies
  for (int l = 0; l < L_; l++) {
    wt_kernel<<<D_, D_, 0, stream>>>(gW1 + (size_t)l * D_ * D_, W1T + (size_t)l * D_ * D_);
    wt_kernel<<<D_, D_, 0, stream>>>(gW2 + (size_t)l * D_ * D_, W2T + (size_t)l * D_ * D_);
  }
  wt_kernel<<<D_, D_, 0, stream>>>(Wn, WnT);

  encode_kernel<<<N_, D_, 0, stream>>>(x, W_enc, b_enc, h0);
  init_kernel<<<N_, D_, 0, stream>>>(h0, bufA);
  zero_i32<<<(M_ * N_ + 255) / 256, 256, 0, stream>>>(anchor, M_ * N_);

  const size_t SM_A   = (size_t)64 * AS_STRIDE * 2;                    // 33792
  const size_t SM_AGG = SM_A + (size_t)(EPG_ * 3 + 64 + 65) * 4 + 64;  // ~40.6 KB
  const size_t SM_MD  = SM_A + (size_t)(64 + 256 + 256) * 4;
  const int NWG = (M_ * N_) / 64;   // 1024

  auto gnn = [&]() {
    for (int l = 0; l < L_; l++) {
      gemm_kernel<<<NWG, 256, SM_AGG, stream>>>(
          bufA, bufB, W1T + (size_t)l * D_ * D_, gb1 + l * D_, eps, l, esrc, edst, 1);
      gemm_kernel<<<NWG, 256, SM_A, stream>>>(
          bufB, bufA, W2T + (size_t)l * D_ * D_, gb2 + l * D_, eps, l, esrc, edst, 0);
    }
  };

  for (int it = 1; it <= NA_; it++) {
    gnn();
    mindist_kernel<<<M_ * B_, 256, SM_MD, stream>>>(bufA, mind);
    anchor_kernel<<<M_ * B_, 64, 0, stream>>>(mind, anchor, it);
    xcur_kernel<<<M_ * N_, D_, 0, stream>>>(h0, anchor, aemb, bufA);
  }
  gnn();

  // hn = relu(h @ W_n2n + b_n2n)
  gemm_kernel<<<NWG, 256, SM_A, stream>>>(bufA, bufB, WnT, bn, eps, 0, esrc, edst, 0);
  pool_kernel<<<M_ * B_, D_, 0, stream>>>(bufB, hg);
  head_kernel<<<B_, 64, 0, stream>>>(hg, Wp, bp, out);
}
